// MultiHeadAttention_19670950215919
// MI455X (gfx1250) — matrix operand order repaired and checked by reading, not compiled, not run on hardware
//
#include <hip/hip_runtime.h>

// ---------------------------------------------------------------------------
// MI455X (gfx1250) causal MHA, WMMA f16-in / f32-acc pipeline, v3.
//
//   pre :  x_h = f16(x);  Wt_in = f16(W_in^T);  Wt_out = f16(W_out^T)
//   1)  :  qkv_h = x_h @ Wt_in^T + b_in   (direct-from-global WMMA GEMM,
//          Q written pre-scaled by 0.125, V written transposed into Vt)
//   2)  :  attn_h = causal flash attention, 128-key blocks; row-sum l kept
//          as a WMMA accumulator (P @ ones), only row-max uses shuffles.
//   3)  :  out = attn_h @ Wt_out^T + b_out  (f32 output)
// ---------------------------------------------------------------------------

typedef __attribute__((ext_vector_type(16))) _Float16 v16h_t;
typedef __attribute__((ext_vector_type(8)))  _Float16 v8h_t;
typedef __attribute__((ext_vector_type(4)))  _Float16 v4h_t;
typedef __attribute__((ext_vector_type(8)))  float    v8f_t;

#define SEQ_LEN   4096
#define D_EMBED   1024
#define N_HEADS   16
#define D_HEAD    64
#define QKV_LD    (3 * D_EMBED)
#define FA_BK     128   // keys per flash iteration

static __device__ __forceinline__ v8f_t wmma_f16(v16h_t a, v16h_t b, v8f_t c) {
  return __builtin_amdgcn_wmma_f32_16x16x32_f16(false, a, false, b, (short)0, c,
                                                false, false);
}

// A-fragment (16x32 f16) from a row-major tile (global or LDS), ld in halves.
// lanes 0-15: rows M=0..15, K = {0..7, 16..23}; lanes 16-31: same rows, K+8.
static __device__ __forceinline__ v16h_t load_a_frag(const _Float16* base,
                                                     int ld, int lane) {
  const int row = lane & 15;
  const int kb  = (lane < 16) ? 0 : 8;
  const _Float16* p = base + (size_t)row * ld + kb;
  v8h_t lo = *(const v8h_t*)(p);
  v8h_t hi = *(const v8h_t*)(p + 16);
  v16h_t f;
#pragma unroll
  for (int i = 0; i < 8; ++i) { f[i] = lo[i]; f[8 + i] = hi[i]; }
  return f;
}

// B-fragment (32x16 f16) from an N-major tile (contiguous K per column).
// lanes 0-15: N=0..15, K = {0..7, 16..23}; lanes 16-31: same N, K+8.
static __device__ __forceinline__ v16h_t load_b_frag(const _Float16* base,
                                                     int ld, int lane) {
  const int n  = lane & 15;
  const int kb = (lane < 16) ? 0 : 8;
  const _Float16* p = base + (size_t)n * ld + kb;
  v8h_t lo = *(const v8h_t*)(p);
  v8h_t hi = *(const v8h_t*)(p + 16);
  v16h_t f;
#pragma unroll
  for (int i = 0; i < 8; ++i) { f[i] = lo[i]; f[8 + i] = hi[i]; }
  return f;
}

// ---------------------------------------------------------------------------
// Pre-pass: f32 -> f16 elementwise (n multiple of 1024).
// ---------------------------------------------------------------------------
__global__ __launch_bounds__(256) void cvt_f16_kernel(
    const float* __restrict__ in, _Float16* __restrict__ out, int n) {
  const int i = (blockIdx.x * 256 + threadIdx.x) * 4;
  if (i < n) {
    float4 v = *(const float4*)(in + i);
    v4h_t h;
    h[0] = (_Float16)v.x; h[1] = (_Float16)v.y;
    h[2] = (_Float16)v.z; h[3] = (_Float16)v.w;
    *(v4h_t*)(out + i) = h;
  }
}

// ---------------------------------------------------------------------------
// Pre-pass: Wt[n][k] = f16(W[k][n]).  W is K x N (f32).  64x64 LDS tile.
// ---------------------------------------------------------------------------
__global__ __launch_bounds__(256) void transpose_cvt_f16_kernel(
    const float* __restrict__ W, _Float16* __restrict__ Wt, int K, int N) {
  __shared__ __align__(16) _Float16 T[64 * 65];
  const int t  = threadIdx.x;
  const int n0 = blockIdx.x * 64;
  const int k0 = blockIdx.y * 64;
  {
    const int kk = t >> 2;
    const int nn = (t & 3) * 16;
    const float4* src = (const float4*)(W + (size_t)(k0 + kk) * N + n0 + nn);
    _Float16* dst = T + kk * 65 + nn;
#pragma unroll
    for (int i = 0; i < 4; ++i) {
      float4 v = src[i];
      dst[i * 4 + 0] = (_Float16)v.x; dst[i * 4 + 1] = (_Float16)v.y;
      dst[i * 4 + 2] = (_Float16)v.z; dst[i * 4 + 3] = (_Float16)v.w;
    }
  }
  __syncthreads();
  {
    const int nn = t >> 2;
    const int kk = (t & 3) * 16;
    v8h_t p0, p1;
#pragma unroll
    for (int j = 0; j < 8; ++j) p0[j] = T[(kk + j) * 65 + nn];
#pragma unroll
    for (int j = 0; j < 8; ++j) p1[j] = T[(kk + 8 + j) * 65 + nn];
    _Float16* dst = Wt + (size_t)(n0 + nn) * K + k0 + kk;
    *(v8h_t*)(dst + 0) = p0;
    *(v8h_t*)(dst + 8) = p1;
  }
}

// ---------------------------------------------------------------------------
// Direct-from-global WMMA GEMM:  acc[M,N] = A[M,K] @ Bt[N,K]^T  (f16 in).
// 128 threads = 4 waves (2x2); wave tile 64x64 => workgroup tile 128x128.
// No LDS, no barriers: fragments stream from L2-resident f16 operands.
// MODE 0: C = acc + bias (f32).
// MODE 1: qkv epilogue: col<1024 -> qkv_h f16 scaled 0.125 (Q),
//         col<2048 -> qkv_h f16 (K), else -> Vt[col-2048][row] f16 (V^T).
// ---------------------------------------------------------------------------
template <int MODE>
__global__ __launch_bounds__(128) void wmma_gemm_dg_kernel(
    const _Float16* __restrict__ A, const _Float16* __restrict__ Bt,
    const float* __restrict__ bias, float* __restrict__ Cf,
    _Float16* __restrict__ qkvh, _Float16* __restrict__ Vt,
    int M, int N, int K) {
  const int t    = threadIdx.x;
  const int lane = t & 31;
  const int wave = t >> 5;
  const int m_w  = blockIdx.y * 128 + (wave >> 1) * 64;
  const int n_w  = blockIdx.x * 128 + (wave & 1) * 64;
  const int nlo  = lane & 15;
  const int rof  = (lane < 16) ? 0 : 8;

  v8f_t acc[4][4];
#pragma unroll
  for (int i = 0; i < 4; ++i)
#pragma unroll
    for (int j = 0; j < 4; ++j) acc[i][j] = (v8f_t){};

  for (int k0 = 0; k0 < K; k0 += 32) {
    if (k0 + 32 < K) {  // light L2 prefetch touch for the next k-slab
      __builtin_prefetch(A + (size_t)(m_w + nlo) * K + k0 + 32, 0, 1);
      __builtin_prefetch(Bt + (size_t)(n_w + nlo) * K + k0 + 32, 0, 1);
    }
    v16h_t a[4], b[4];
#pragma unroll
    for (int i = 0; i < 4; ++i)
      a[i] = load_a_frag(A + (size_t)(m_w + 16 * i) * K + k0, K, lane);
#pragma unroll
    for (int j = 0; j < 4; ++j)
      b[j] = load_b_frag(Bt + (size_t)(n_w + 16 * j) * K + k0, K, lane);
#pragma unroll
    for (int i = 0; i < 4; ++i)
#pragma unroll
      for (int j = 0; j < 4; ++j) acc[i][j] = wmma_f16(a[i], b[j], acc[i][j]);
  }

#pragma unroll
  for (int sr = 0; sr < 4; ++sr) {
#pragma unroll
    for (int sc = 0; sc < 4; ++sc) {
      const int col  = n_w + sc * 16 + nlo;
      const int row0 = m_w + sr * 16 + rof;
      const float bv = bias[col];
      if (MODE == 0) {
#pragma unroll
        for (int r = 0; r < 8; ++r)
          Cf[(size_t)(row0 + r) * N + col] = acc[sr][sc][r] + bv;
      } else {
        if (col < D_EMBED) {  // Q: fold 1/sqrt(dh)=0.125 (exact pow2 in f16)
#pragma unroll
          for (int r = 0; r < 8; ++r)
            qkvh[(size_t)(row0 + r) * QKV_LD + col] =
                (_Float16)((acc[sr][sc][r] + bv) * 0.125f);
        } else if (col < 2 * D_EMBED) {  // K
#pragma unroll
          for (int r = 0; r < 8; ++r)
            qkvh[(size_t)(row0 + r) * QKV_LD + col] =
                (_Float16)(acc[sr][sc][r] + bv);
        } else {  // V, transposed: Vt[d_global][key], rows contiguous per lane
          v8h_t pk;
#pragma unroll
          for (int r = 0; r < 8; ++r) pk[r] = (_Float16)(acc[sr][sc][r] + bv);
          *(v8h_t*)(Vt + (size_t)(col - 2 * D_EMBED) * SEQ_LEN + row0) = pk;
        }
      }
    }
  }
}

// ---------------------------------------------------------------------------
// Causal flash attention, barrier-free, 128-key blocks.
// Grid: (SEQ/64, N_HEADS), 128 threads = 4 independent waves; each wave owns
// 16 query rows. Row-sum l is a WMMA accumulator tile (P @ ones); only the
// row-max reduction uses cross-lane shuffles.
// ---------------------------------------------------------------------------
__global__ __launch_bounds__(128) void flash_attn_causal_kernel(
    const _Float16* __restrict__ qkvh, const _Float16* __restrict__ Vt,
    _Float16* __restrict__ attnh) {
  __shared__ __align__(16) _Float16 Ps[4][16 * FA_BK];  // per-wave P, ld 128

  const int t     = threadIdx.x;
  const int lane  = t & 31;
  const int wave  = t >> 5;
  const int h     = blockIdx.y;
  const int qb    = blockIdx.x;
  const int qbase = qb * 64;

  const int nlo = lane & 15;
  const int rof = (lane < 16) ? 0 : 8;

  // Q fragments (already scaled by 0.125 at the GEMM1 epilogue).
  v16h_t qf[2];
#pragma unroll
  for (int kk = 0; kk < 2; ++kk)
    qf[kk] = load_a_frag(qkvh + (size_t)(qbase + wave * 16) * QKV_LD +
                             h * D_HEAD + kk * 32,
                         QKV_LD, lane);

  // Constant ones B-fragment: P @ ones = per-row sum, column-replicated.
  v16h_t ones;
#pragma unroll
  for (int i = 0; i < 16; ++i) ones[i] = (_Float16)1.0f;

  v8f_t o[4] = {(v8f_t){}, (v8f_t){}, (v8f_t){}, (v8f_t){}};
  v8f_t l_acc = {};
  float mi[8];
#pragma unroll
  for (int r = 0; r < 8; ++r) mi[r] = -3.0e30f;

  const int nkb = (qbase + 64 + FA_BK - 1) >> 7;  // 128-key blocks needed
  for (int kb = 0; kb < nkb; ++kb) {
    const int kbase = kb * FA_BK;

    // --- Scores: S = Q (16x64) x K^T (64x128); 8 N-tiles x 2 K-steps ---
    v8f_t st[8];
#pragma unroll
    for (int nt = 0; nt < 8; ++nt) {
      const _Float16* kp = qkvh + (size_t)(kbase + nt * 16) * QKV_LD +
                           D_EMBED + h * D_HEAD;
      v8f_t s = {};
      s = wmma_f16(qf[0], load_b_frag(kp + 0, QKV_LD, lane), s);
      s = wmma_f16(qf[1], load_b_frag(kp + 32, QKV_LD, lane), s);
      st[nt] = s;
    }

    // --- Causal mask (only the last key block can cross the diagonal) ---
    if (kb == nkb - 1) {
#pragma unroll
      for (int nt = 0; nt < 8; ++nt) {
        const int col = kbase + nt * 16 + nlo;
#pragma unroll
        for (int r = 0; r < 8; ++r) {
          const int row = qbase + wave * 16 + rof + r;
          if (col > row) st[nt][r] = -3.0e30f;
        }
      }
    }

    // --- Row max: tile-local fmax chain + 16-lane butterfly ---
    float mnew[8];
#pragma unroll
    for (int r = 0; r < 8; ++r) {
      float mv = mi[r];
#pragma unroll
      for (int nt = 0; nt < 8; ++nt) mv = fmaxf(mv, st[nt][r]);
#pragma unroll
      for (int off = 1; off < 16; off <<= 1)
        mv = fmaxf(mv, __shfl_xor(mv, off, 32));
      mnew[r] = mv;
    }

    // --- exp + stage P (f16) into wave-private LDS ---
#pragma unroll
    for (int nt = 0; nt < 8; ++nt) {
#pragma unroll
      for (int r = 0; r < 8; ++r) {
        const float p = __expf(st[nt][r] - mnew[r]);
        Ps[wave][(rof + r) * FA_BK + nt * 16 + nlo] = (_Float16)p;
      }
    }

    // --- Rescale running O and l by alpha = exp(m_old - m_new) ---
#pragma unroll
    for (int r = 0; r < 8; ++r) {
      const float alpha = __expf(mi[r] - mnew[r]);
      mi[r] = mnew[r];
      l_acc[r] *= alpha;
      o[0][r] *= alpha; o[1][r] *= alpha; o[2][r] *= alpha; o[3][r] *= alpha;
    }

    // --- P (C-layout) -> A fragments via wave-private LDS (DS in-order) ---
    v16h_t pa[4];
#pragma unroll
    for (int kk = 0; kk < 4; ++kk)
      pa[kk] = load_a_frag(&Ps[wave][kk * 32], FA_BK, lane);

    // --- l += P @ ones  (row sums on the matrix pipe) ---
#pragma unroll
    for (int kk = 0; kk < 4; ++kk) l_acc = wmma_f16(pa[kk], ones, l_acc);

    // --- O += P (16x128) x V (128x64); V fragments direct from Vt ---
    const _Float16* vb = Vt + (size_t)(h * D_HEAD) * SEQ_LEN + kbase;
#pragma unroll
    for (int dt = 0; dt < 4; ++dt) {
#pragma unroll
      for (int kk = 0; kk < 4; ++kk)
        o[dt] = wmma_f16(
            pa[kk],
            load_b_frag(vb + (size_t)(dt * 16) * SEQ_LEN + kk * 32, SEQ_LEN,
                        lane),
            o[dt]);
    }
  }

  // --- Epilogue: attn_h[row, h*64+d] = (f16)(O / l) ---
#pragma unroll
  for (int r = 0; r < 8; ++r) {
    const int row = qbase + wave * 16 + rof + r;
    const float inv = 1.0f / l_acc[r];
    _Float16* op = attnh + (size_t)row * D_EMBED + h * D_HEAD;
    op[0  + nlo] = (_Float16)(o[0][r] * inv);
    op[16 + nlo] = (_Float16)(o[1][r] * inv);
    op[32 + nlo] = (_Float16)(o[2][r] * inv);
    op[48 + nlo] = (_Float16)(o[3][r] * inv);
  }
}

// ---------------------------------------------------------------------------
extern "C" void kernel_launch(void* const* d_in, const int* in_sizes, int n_in,
                              void* d_out, int out_size, void* d_ws, size_t ws_size,
                              hipStream_t stream) {
  const float* x     = (const float*)d_in[0];  // [4096, 1024]
  const float* W_in  = (const float*)d_in[1];  // [1024, 3072]
  const float* b_in  = (const float*)d_in[2];  // [3072]
  const float* W_out = (const float*)d_in[3];  // [1024, 1024]
  const float* b_out = (const float*)d_in[4];  // [1024]
  float* out = (float*)d_out;                  // [4096, 1024] f32

  // Workspace carve (all f16, all 16B-aligned): total 56 MB.
  _Float16* x_h    = (_Float16*)d_ws;                       // 4096*1024
  _Float16* Wt_in  = x_h    + (size_t)SEQ_LEN * D_EMBED;    // 3072*1024
  _Float16* Wt_out = Wt_in  + (size_t)QKV_LD * D_EMBED;     // 1024*1024
  _Float16* qkv_h  = Wt_out + (size_t)D_EMBED * D_EMBED;    // 4096*3072 (Q,K used)
  _Float16* Vt     = qkv_h  + (size_t)SEQ_LEN * QKV_LD;     // 1024*4096 (V^T)
  _Float16* attn_h = Vt     + (size_t)D_EMBED * SEQ_LEN;    // 4096*1024

  // --- Pre-pass conversions (one-time layout/precision fixups) ---
  {
    const int n = SEQ_LEN * D_EMBED;
    cvt_f16_kernel<<<dim3(n / 1024), dim3(256), 0, stream>>>(x, x_h, n);
  }
  transpose_cvt_f16_kernel<<<dim3(QKV_LD / 64, D_EMBED / 64), dim3(256), 0,
                             stream>>>(W_in, Wt_in, D_EMBED, QKV_LD);
  transpose_cvt_f16_kernel<<<dim3(D_EMBED / 64, D_EMBED / 64), dim3(256), 0,
                             stream>>>(W_out, Wt_out, D_EMBED, D_EMBED);

  // --- 1) qkv = x @ W_in + b_in  (Q scaled, V transposed at the epilogue) ---
  wmma_gemm_dg_kernel<1><<<dim3(QKV_LD / 128, SEQ_LEN / 128), dim3(128), 0,
                           stream>>>(x_h, Wt_in, b_in, nullptr, qkv_h, Vt,
                                     SEQ_LEN, QKV_LD, D_EMBED);

  // --- 2) causal flash attention ---
  flash_attn_causal_kernel<<<dim3(SEQ_LEN / 64, N_HEADS), dim3(128), 0,
                             stream>>>(qkv_h, Vt, attn_h);

  // --- 3) out = attn @ W_out + b_out ---
  wmma_gemm_dg_kernel<0><<<dim3(D_EMBED / 128, SEQ_LEN / 128), dim3(128), 0,
                           stream>>>(attn_h, Wt_out, b_out, out, nullptr,
                                     nullptr, SEQ_LEN, D_EMBED, D_EMBED);
}
